// MultiHeadAttention_20418274525433
// MI455X (gfx1250) — matrix operand order repaired and checked by reading, not compiled, not run on hardware
//
#include <hip/hip_runtime.h>
#include <hip/hip_bf16.h>

// CDNA5 / gfx1250, wave32. Matrix ops via v_wmma_f32_16x16x32_bf16.
// K-tile staging via global_load_async_to_lds_b128 (ASYNCcnt path).

typedef __attribute__((ext_vector_type(16))) __bf16 v16bf;
typedef __attribute__((ext_vector_type(8)))  float  v8f;

#define EMB      1024
#define HEADS    16
#define DKDIM    64
#define DVDIM    64
#define BATCH    4
#define SQLEN    2048
#define SKVLEN   1024   // rank-0 keeps first half of 2048
#define SKV_FULL 2048

union AFrag { v16bf v; unsigned u[8]; unsigned short s[16]; };
union CFrag { v8f v; float f[8]; };

// native bf16 converts (backend legalizes fptrunc f32->bf16, RNE)
__device__ __forceinline__ unsigned short bf16c(float x) {
  union { unsigned short u; __bf16 h; } r; r.h = (__bf16)x; return r.u;
}
__device__ __forceinline__ unsigned pkbf(float a, float b) {
  union { unsigned u; __bf16 h[2]; } r; r.h[0] = (__bf16)a; r.h[1] = (__bf16)b;
  return r.u;
}

// 16-bit A matrix 16x32 layout (ISA 7.12.2): per-lane element pair p -> K index
__device__ __forceinline__ int kpairA(int p, int hi) {
  return (p < 4) ? ((hi ? 8 : 0) + 2 * p) : ((hi ? 24 : 16) + 2 * (p - 4));
}
// 16-bit B matrix 32x16 layout: same per-lane K order as A (pair p -> K index)
__device__ __forceinline__ int kpairB(int p, int hi) {
  return (p < 4) ? ((hi ? 8 : 0) + 2 * p) : ((hi ? 24 : 16) + 2 * (p - 4));
}

__device__ __forceinline__ v8f wmma_bf16(const AFrag& a, const AFrag& b, v8f c) {
  return __builtin_amdgcn_wmma_f32_16x16x32_bf16(false, a.v, false, b.v,
                                                 (short)0, c, false, false);
}

// ---------------------------------------------------------------------------
// Projection GEMM: Y[M,1024](bf16) = X[M(in),1024](f32) * W[1024,1024]^T + b
// 32x64 output tile per wave (2 A-frags x 4 B-frags = 8 WMMA / K-step of 32).
// kvhalf=1 maps output row m -> input row (m/1024)*2048 + (m%1024).
// ---------------------------------------------------------------------------
__global__ __launch_bounds__(256)
void proj_kernel(const float* __restrict__ X, const float* __restrict__ W,
                 const float* __restrict__ bias, unsigned short* __restrict__ Y,
                 int Mrows, int kvhalf) {
  __shared__ unsigned short sA[8][32 * 32];   // per-wave A tile [m][k]
  __shared__ unsigned short sB[8][64 * 32];   // per-wave B tile [n][k]
  const int lane   = threadIdx.x & 31;
  const int wave   = threadIdx.x >> 5;
  const int laneLo = lane & 15;
  const int hi     = lane >> 4;
  const int colTiles = EMB / 64;                  // 16
  const int tile = blockIdx.x * 8 + wave;
  if (tile >= (Mrows / 32) * colTiles) return;
  const int row0 = (tile / colTiles) * 32;
  const int col0 = (tile % colTiles) * 64;
  unsigned short* As = sA[wave];
  unsigned short* Bs = sB[wave];

  CFrag c[8];
#pragma unroll
  for (int i = 0; i < 8; ++i)
#pragma unroll
    for (int r = 0; r < 8; ++r) c[i].f[r] = 0.f;

  for (int k0 = 0; k0 < EMB; k0 += 32) {
    // A tile 32x32 f32 -> bf16 (float4 loads, packed 64-bit LDS stores)
#pragma unroll
    for (int i = 0; i < 8; ++i) {
      const int idx = i * 32 + lane;             // 256 float4 slots
      const int row = idx >> 3;                  // 0..31
      const int seg = idx & 7;                   // 8 float4 per row
      const int rowA  = row0 + row;
      const int inRow = kvhalf ? ((rowA >> 10) * SKV_FULL + (rowA & 1023)) : rowA;
      const float4 f = *(const float4*)&X[(size_t)inRow * EMB + k0 + seg * 4];
      uint2 pk; pk.x = pkbf(f.x, f.y); pk.y = pkbf(f.z, f.w);
      *(uint2*)&As[row * 32 + seg * 4] = pk;
    }
    // B tile 64x32
#pragma unroll
    for (int i = 0; i < 16; ++i) {
      const int idx = i * 32 + lane;             // 512 float4 slots
      const int row = idx >> 3;                  // 0..63
      const int seg = idx & 7;
      const float4 f = *(const float4*)&W[(size_t)(col0 + row) * EMB + k0 + seg * 4];
      uint2 pk; pk.x = pkbf(f.x, f.y); pk.y = pkbf(f.z, f.w);
      *(uint2*)&Bs[row * 32 + seg * 4] = pk;
    }
    asm volatile("s_wait_dscnt 0" ::: "memory");
    AFrag a[2], b[4];
#pragma unroll
    for (int p = 0; p < 8; ++p) {
      const int ka = kpairA(p, hi);
      const int kb = kpairB(p, hi);
      a[0].u[p] = *(const unsigned*)&As[laneLo * 32 + ka];
      a[1].u[p] = *(const unsigned*)&As[(16 + laneLo) * 32 + ka];
      b[0].u[p] = *(const unsigned*)&Bs[laneLo * 32 + kb];
      b[1].u[p] = *(const unsigned*)&Bs[(16 + laneLo) * 32 + kb];
      b[2].u[p] = *(const unsigned*)&Bs[(32 + laneLo) * 32 + kb];
      b[3].u[p] = *(const unsigned*)&Bs[(48 + laneLo) * 32 + kb];
    }
#pragma unroll
    for (int mi = 0; mi < 2; ++mi)
#pragma unroll
      for (int ni = 0; ni < 4; ++ni)
        c[mi * 4 + ni].v = wmma_bf16(a[mi], b[ni], c[mi * 4 + ni].v);
  }

#pragma unroll
  for (int ni = 0; ni < 4; ++ni) {
    const float bv = bias[col0 + ni * 16 + laneLo];
#pragma unroll
    for (int mi = 0; mi < 2; ++mi)
#pragma unroll
      for (int r = 0; r < 8; ++r) {
        const int m = r + 8 * hi;
        Y[(size_t)(row0 + mi * 16 + m) * EMB + col0 + ni * 16 + laneLo] =
            bf16c(c[mi * 4 + ni].f[r] + bv);
      }
  }
}

// ---------------------------------------------------------------------------
// Flash attention: one WG per (b, h, 128-q block). 8 waves x 16 q-rows.
// K tile (32x64 bf16 = 4KB) staged via async-to-LDS; V transposed via VALU;
// next tile prefetched. Online softmax in f32 (v_exp_f32), 16-wide shfl.
// ---------------------------------------------------------------------------
__global__ __launch_bounds__(256)
void attn_kernel(const unsigned short* __restrict__ Qb,
                 const unsigned short* __restrict__ Kb,
                 const unsigned short* __restrict__ Vb,
                 const unsigned char* __restrict__ mask,   // bool [B,SQ,SKV]
                 float* __restrict__ out) {
  __shared__ unsigned short Ks[32 * 64];   // [kv][dk]
  __shared__ unsigned short Vt[64 * 32];   // [dv][kv] (transposed)
  __shared__ unsigned short Ps[8][16 * 32];

  const int t      = threadIdx.x;
  const int lane   = t & 31;
  const int wave   = t >> 5;
  const int laneLo = lane & 15;
  const int hi     = lane >> 4;

  const int blk  = blockIdx.x;
  const int qblk = blk & 15;
  const int h    = (blk >> 4) & 15;
  const int b    = blk >> 8;
  const int row0 = qblk * 128 + wave * 16;

  // Q A-fragments (dk 0..31 and 32..63), loaded once
  AFrag a0, a1;
  {
    const unsigned short* qrow =
        Qb + (size_t)(b * SQLEN + row0 + laneLo) * EMB + h * DKDIM;
#pragma unroll
    for (int p = 0; p < 8; ++p) {
      const int k = kpairA(p, hi);
      a0.u[p] = *(const unsigned*)&qrow[k];
      a1.u[p] = *(const unsigned*)&qrow[32 + k];
    }
  }

  float mrow[8], lrow[8];
  CFrag o[4];
#pragma unroll
  for (int r = 0; r < 8; ++r) { mrow[r] = -3.0e38f; lrow[r] = 0.f; }
#pragma unroll
  for (int c2 = 0; c2 < 4; ++c2)
#pragma unroll
    for (int r = 0; r < 8; ++r) o[c2].f[r] = 0.f;

  const float scale = 0.125f;  // 1/sqrt(64)
  const float LOG2E = 1.4426950408889634f;

  for (int jt = 0; jt < SKVLEN / 32; ++jt) {
    const int kv0 = jt * 32;
    __syncthreads();

    // --- K tile: async copy global -> LDS, 16B per thread (ASYNCcnt path)
    {
      const int row = t >> 3;                 // 0..31 kv
      const int seg = t & 7;                  // 8 x 16B per 64-bf16 row
      const unsigned ldsOff = (unsigned)(size_t)&Ks[row * 64 + seg * 8];
      const unsigned long long ga = (unsigned long long)(size_t)
          &Kb[(size_t)(b * SKVLEN + kv0 + row) * EMB + h * DKDIM + seg * 8];
      asm volatile("global_load_async_to_lds_b128 %0, %1, off"
                   :: "v"(ldsOff), "v"(ga) : "memory");
      // prefetch next KV tile while this one is consumed
      if (jt + 1 < SKVLEN / 32) {
        __builtin_prefetch(
            &Kb[(size_t)(b * SKVLEN + kv0 + 32 + row) * EMB + h * DKDIM + seg * 8], 0, 0);
        __builtin_prefetch(
            &Vb[(size_t)(b * SKVLEN + kv0 + 32 + row) * EMB + h * DVDIM + seg * 8], 0, 0);
      }
    }
    // --- V tile: regular loads, transposed into LDS [dv][kv]
#pragma unroll
    for (int i = 0; i < 4; ++i) {
      const int idx = t + i * 256;            // 1024 uint slots
      const int row = idx >> 5;               // kv
      const int cp  = idx & 31;               // uint column (2 dv)
      const unsigned vd = *(const unsigned*)
          &Vb[(size_t)(b * SKVLEN + kv0 + row) * EMB + h * DVDIM + cp * 2];
      Vt[(cp * 2) * 32 + row]     = (unsigned short)(vd & 0xFFFFu);
      Vt[(cp * 2 + 1) * 32 + row] = (unsigned short)(vd >> 16);
    }
    asm volatile("s_wait_asynccnt 0" ::: "memory");
    __syncthreads();

    // S = Q K^T : two 16x16 tiles (kv columns 0-15 and 16-31)
    AFrag bL0, bL1, bR0, bR1;
#pragma unroll
    for (int p = 0; p < 8; ++p) {
      const int k = kpairB(p, hi);
      bL0.u[p] = *(const unsigned*)&Ks[laneLo * 64 + k];
      bL1.u[p] = *(const unsigned*)&Ks[laneLo * 64 + 32 + k];
      bR0.u[p] = *(const unsigned*)&Ks[(16 + laneLo) * 64 + k];
      bR1.u[p] = *(const unsigned*)&Ks[(16 + laneLo) * 64 + 32 + k];
    }
    CFrag sL, sR;
#pragma unroll
    for (int r = 0; r < 8; ++r) { sL.f[r] = 0.f; sR.f[r] = 0.f; }
    sL.v = wmma_bf16(a0, bL0, sL.v);
    sL.v = wmma_bf16(a1, bL1, sL.v);
    sR.v = wmma_bf16(a0, bR0, sR.v);
    sR.v = wmma_bf16(a1, bR1, sR.v);

    // mask + scale + online softmax (per output row)
#pragma unroll
    for (int r = 0; r < 8; ++r) {
      const int m = r + 8 * hi;
      const size_t mbase = (size_t)(b * SQLEN + row0 + m) * SKVLEN + kv0;
      float vL = sL.f[r] * scale;
      float vR = sR.f[r] * scale;
      if (mask[mbase + laneLo])      vL = -1.0e9f;   // mask True -> -1e9
      if (mask[mbase + 16 + laneLo]) vR = -1.0e9f;
      float tmax = fmaxf(vL, vR);
#pragma unroll
      for (int off = 8; off >= 1; off >>= 1)
        tmax = fmaxf(tmax, __shfl_xor(tmax, off, 16));
      const float mnew  = fmaxf(mrow[r], tmax);
      const float alpha = exp2f((mrow[r] - mnew) * LOG2E);
      const float pL = exp2f((vL - mnew) * LOG2E);
      const float pR = exp2f((vR - mnew) * LOG2E);
      float rs = pL + pR;
#pragma unroll
      for (int off = 8; off >= 1; off >>= 1)
        rs += __shfl_xor(rs, off, 16);
      lrow[r] = lrow[r] * alpha + rs;
      mrow[r] = mnew;
#pragma unroll
      for (int c2 = 0; c2 < 4; ++c2) o[c2].f[r] *= alpha;
      Ps[wave][m * 32 + laneLo]      = bf16c(pL);
      Ps[wave][m * 32 + 16 + laneLo] = bf16c(pR);
    }
    asm volatile("s_wait_dscnt 0" ::: "memory");

    // O += P * V   (P: 16x32 A-frag, V: 32x16 B-frags for 4 dv chunks)
    AFrag pa;
#pragma unroll
    for (int p = 0; p < 8; ++p)
      pa.u[p] = *(const unsigned*)&Ps[wave][laneLo * 32 + kpairA(p, hi)];
#pragma unroll
    for (int c2 = 0; c2 < 4; ++c2) {
      AFrag vb;
#pragma unroll
      for (int p = 0; p < 8; ++p)
        vb.u[p] = *(const unsigned*)&Vt[(c2 * 16 + laneLo) * 32 + kpairB(p, hi)];
      o[c2].v = wmma_bf16(pa, vb, o[c2].v);
    }
  }

  // normalize + write f32 output [B, SQ, H*DV]
#pragma unroll
  for (int r = 0; r < 8; ++r) {
    const int m = r + 8 * hi;
    const float inv = (lrow[r] > 0.f) ? (1.0f / lrow[r]) : 0.f;
    float* orow = out + (size_t)(b * SQLEN + row0 + m) * (HEADS * DVDIM) + h * DVDIM;
#pragma unroll
    for (int c2 = 0; c2 < 4; ++c2)
      orow[c2 * 16 + laneLo] = o[c2].f[r] * inv;
  }
}

// ---------------------------------------------------------------------------
extern "C" void kernel_launch(void* const* d_in, const int* in_sizes, int n_in,
                              void* d_out, int out_size, void* d_ws, size_t ws_size,
                              hipStream_t stream) {
  const float* x_q  = (const float*)d_in[0];
  const float* x_kv = (const float*)d_in[1];
  const unsigned char* mask = (const unsigned char*)d_in[2]; // bool array
  const float* w_q_w = (const float*)d_in[3];
  const float* w_q_b = (const float*)d_in[4];
  const float* w_k_w = (const float*)d_in[5];
  const float* w_k_b = (const float*)d_in[6];
  const float* w_v_w = (const float*)d_in[7];
  const float* w_v_b = (const float*)d_in[8];
  float* out = (float*)d_out;

  unsigned short* Qb = (unsigned short*)d_ws;                 // [8192,1024] bf16
  unsigned short* Kb = Qb + (size_t)BATCH * SQLEN * EMB;      // [4096,1024] bf16
  unsigned short* Vb = Kb + (size_t)BATCH * SKVLEN * EMB;     // [4096,1024] bf16

  {
    const int Mrows = BATCH * SQLEN;                          // 8192
    const int tiles = (Mrows / 32) * (EMB / 64);              // 4096
    proj_kernel<<<tiles / 8, 256, 0, stream>>>(x_q, w_q_w, w_q_b, Qb, Mrows, 0);
  }
  {
    const int Mrows = BATCH * SKVLEN;                         // 4096 (first-half kv)
    const int tiles = (Mrows / 32) * (EMB / 64);              // 2048
    proj_kernel<<<tiles / 8, 256, 0, stream>>>(x_kv, w_k_w, w_k_b, Kb, Mrows, 1);
    proj_kernel<<<tiles / 8, 256, 0, stream>>>(x_kv, w_v_w, w_v_b, Vb, Mrows, 1);
  }
  attn_kernel<<<BATCH * HEADS * (SQLEN / 128), 256, 0, stream>>>(Qb, Kb, Vb, mask, out);
}
